// CBOWModule_15281493639247
// MI455X (gfx1250) — matrix operand order repaired and checked by reading, not compiled, not run on hardware
//
#include <hip/hip_runtime.h>
#include <hip/hip_bf16.h>

// ---- CDNA5 vector types ----
typedef _Float16 h16 __attribute__((ext_vector_type(16)));
typedef _Float16 h8  __attribute__((ext_vector_type(8)));
typedef _Float16 h4  __attribute__((ext_vector_type(4)));
typedef float    f8  __attribute__((ext_vector_type(8)));
typedef float    f4v __attribute__((ext_vector_type(4)));

static constexpr int EMBED = 128;
static constexpr int CW    = 10;    // context window (target gather width)
static constexpr int NK    = 10;    // negative samples
static constexpr int NJ    = 11;    // 1 ctx + 10 negs
static constexpr int TB    = 16;    // batch tile per workgroup (one WMMA M/N)
static constexpr int BATCH = 16384;
static constexpr int ASTRIDE = 136; // f16 row stride in LDS: 272B -> conflict-free frags

// One workgroup (256 thr = 8 waves) handles TB=16 batch elements:
//  Phase 1: gather 21 rows per b (coalesced wave-wide float4), mean targets,
//           convert to f16, stage tiles in LDS.
//  Phase 2: v_wmma_f32_16x16x32_f16 over K=128 (4 chunks), 11 j-vectors spread
//           across the 8 waves; extract diagonal = per-b scores.
//  Phase 3: clip + softplus, per-block partial sum to d_ws (deterministic).
__global__ __launch_bounds__(256) void cbow_tile_kernel(
    const float* __restrict__ tgt_tab,
    const float* __restrict__ ctx_tab,
    const int*   __restrict__ context,
    const int*   __restrict__ target,
    const int*   __restrict__ negatives,
    float* __restrict__ partials)
{
    __shared__ __align__(32) _Float16 As[TB * ASTRIDE];        // trg tile (f16)
    __shared__ __align__(32) _Float16 Vs[TB * NJ * ASTRIDE];   // ctx+neg vectors (f16)
    __shared__ float Ss[NJ * TB];                              // scores S[j][b]
    __shared__ float Ls[TB];                                   // per-b loss

    const int tid  = threadIdx.x;
    const int lane = tid & 31;
    const int wave = tid >> 5;
    const int b0   = blockIdx.x * TB;

    // ---------------- Phase 1: gather + stage ----------------
    // wave w owns local batch rows 2w and 2w+1; 32 lanes x float4 = one 512B row.
    #pragma unroll
    for (int r = 0; r < 2; ++r) {
        const int lb = wave * 2 + r;
        const int gb = b0 + lb;

        f4v acc = {0.f, 0.f, 0.f, 0.f};
        #pragma unroll
        for (int i = 0; i < CW; ++i) {
            const int row = target[gb * CW + i];
            acc += *(const f4v*)(tgt_tab + (size_t)row * EMBED + lane * 4);
        }
        acc *= 0.1f; // mean over window
        *(h4*)(As + lb * ASTRIDE + lane * 4) = __builtin_convertvector(acc, h4);

        #pragma unroll
        for (int j = 0; j < NJ; ++j) {
            const int row = (j == 0) ? context[gb] : negatives[gb * NK + (j - 1)];
            const f4v v = *(const f4v*)(ctx_tab + (size_t)row * EMBED + lane * 4);
            *(h4*)(Vs + (lb * NJ + j) * ASTRIDE + lane * 4) = __builtin_convertvector(v, h4);
        }
    }
    __syncthreads();

    // ---------------- Phase 2: WMMA scores ----------------
    const int m    = lane & 15;   // A-row / B-col handled by this lane
    const int hsel = lane >> 4;   // K-half select per ISA fragment layout

    auto score_j = [&](int j) {
        f8 c = {0.f, 0.f, 0.f, 0.f, 0.f, 0.f, 0.f, 0.f};
        #pragma unroll
        for (int kc = 0; kc < 4; ++kc) {
            // A fragment (16x32 f16): lanes<16 hold K {0-7,16-23}, lanes>=16 {8-15,24-31}
            const int abase = m * ASTRIDE + kc * 32 + hsel * 8;
            h8 alo = *(const h8*)(As + abase);        // 16B aligned
            h8 ahi = *(const h8*)(As + abase + 16);
            h16 a = __builtin_shufflevector(alo, ahi,
                     0,1,2,3,4,5,6,7,8,9,10,11,12,13,14,15);
            // B fragment (32x16 f16): lane = column m, same K order as A: lanes<16 {0-7,16-23}, lanes>=16 {8-15,24-31}
            const int bbase = (m * NJ + j) * ASTRIDE + kc * 32 + hsel * 8;
            h8 blo = *(const h8*)(Vs + bbase);
            h8 bhi = *(const h8*)(Vs + bbase + 16);
            h16 b = __builtin_shufflevector(blo, bhi,
                     0,1,2,3,4,5,6,7,8,9,10,11,12,13,14,15);
            c = __builtin_amdgcn_wmma_f32_16x16x32_f16(
                    false, a, false, b, (short)0, c, false, false);
        }
        // Diagonal extraction: C vgpr r, lanes0-15 -> (M=r,N=lane); lanes16-31 -> (M=8+r,N=lane-16)
        // diag[m<8]  at lane=m,    comp=m
        // diag[m>=8] at lane=m+16, comp=m-8  (lanes 24..31)
        const unsigned rr = lane & 7;
        float s01 = (rr & 1) ? c[1] : c[0];
        float s23 = (rr & 1) ? c[3] : c[2];
        float s45 = (rr & 1) ? c[5] : c[4];
        float s67 = (rr & 1) ? c[7] : c[6];
        float s0123 = (rr & 2) ? s23 : s01;
        float s4567 = (rr & 2) ? s67 : s45;
        float dv = (rr & 4) ? s4567 : s0123;
        if (lane < 8)        Ss[j * TB + lane] = dv;
        else if (lane >= 24) Ss[j * TB + (lane - 16)] = dv;
    };

    score_j(wave);                       // j = 0..7
    if (wave < 3) score_j(8 + wave);     // j = 8..10 (wave-uniform branch, EXEC stays all-1)
    __syncthreads();

    // ---------------- Phase 3: loss + block partial ----------------
    if (tid < TB) {
        float s = Ss[0 * TB + tid];
        s = fminf(fmaxf(s, -10.f), 10.f);
        float loss = log1pf(__expf(-s));            // -log_sigmoid(pos)
        #pragma unroll
        for (int j = 1; j < NJ; ++j) {
            float sn = Ss[j * TB + tid];
            sn = fminf(fmaxf(sn, -10.f), 10.f);
            loss += log1pf(__expf(sn));             // -log_sigmoid(-neg)
        }
        Ls[tid] = loss;
    }
    __syncthreads();
    if (tid == 0) {
        float sum = 0.f;
        #pragma unroll
        for (int i = 0; i < TB; ++i) sum += Ls[i];
        partials[blockIdx.x] = sum;
    }
}

// Deterministic final reduction: fixed-order strided sums + LDS tree.
__global__ __launch_bounds__(256) void cbow_reduce_kernel(
    const float* __restrict__ partials, float* __restrict__ out, int n)
{
    __shared__ float buf[256];
    float s = 0.f;
    for (int i = threadIdx.x; i < n; i += 256) s += partials[i];
    buf[threadIdx.x] = s;
    __syncthreads();
    for (int ofs = 128; ofs > 0; ofs >>= 1) {
        if ((int)threadIdx.x < ofs) buf[threadIdx.x] += buf[threadIdx.x + ofs];
        __syncthreads();
    }
    if (threadIdx.x == 0) out[0] = buf[0] * (1.0f / (float)BATCH);
}

extern "C" void kernel_launch(void* const* d_in, const int* in_sizes, int n_in,
                              void* d_out, int out_size, void* d_ws, size_t ws_size,
                              hipStream_t stream)
{
    (void)in_sizes; (void)n_in; (void)out_size; (void)ws_size;
    const float* tgt_tab   = (const float*)d_in[0];  // target_table  [VOCAB, 128] f32
    const float* ctx_tab   = (const float*)d_in[1];  // context_table [VOCAB, 128] f32
    const int*   context   = (const int*)d_in[2];    // [B]
    const int*   target    = (const int*)d_in[3];    // [B, 10]
    const int*   negatives = (const int*)d_in[4];    // [B, 10]
    float*       out       = (float*)d_out;          // scalar f32
    float*       partials  = (float*)d_ws;           // 1024 floats

    const int nblocks = BATCH / TB;  // 1024
    cbow_tile_kernel<<<nblocks, 256, 0, stream>>>(
        tgt_tab, ctx_tab, context, target, negatives, partials);
    cbow_reduce_kernel<<<1, 256, 0, stream>>>(partials, out, nblocks);
}
